// trajectory2seq_30468497997930
// MI455X (gfx1250) — matrix operand order repaired and checked by reading, not compiled, not run on hardware
//
#include <hip/hip_runtime.h>

constexpr int kH  = 1024;     // hidden
constexpr int kH3 = 3 * kH;   // gates
constexpr int kB  = 256;      // batch
constexpr int kS  = 512;      // encoder seq len
constexpr int kIN = 64;       // encoder input feature dim
constexpr int kV  = 512;      // vocab
constexpr int kT  = 64;       // decoder steps

typedef __bf16 bf16;
typedef __attribute__((ext_vector_type(16))) __bf16 v16bf;
typedef __attribute__((ext_vector_type(8)))  __bf16 v8bf;
typedef __attribute__((ext_vector_type(8)))  float  v8f;

union ABFrag { v16bf v; v8bf h[2]; };

// ---- A fragment: 16x32 bf16, row-major source, lane m = lane&15,
//      K swizzle: ks = (lane>>4)*8 ; chunks [k0, k0+8) and [k0+16, k0+24)
__device__ __forceinline__ v16bf load_a16(const bf16* rowp, int k0) {
  ABFrag f;
  f.h[0] = *(const v8bf*)(rowp + k0);
  f.h[1] = *(const v8bf*)(rowp + k0 + 16);
  return f.v;
}

// ---- B fragment: 32x16 bf16 from pre-transposed weights Wt[K][ldn],
//      lane n = lane&15, K swizzle matches A: [kb, kb+8) and [kb+16, kb+24), kb = (lane>>4)*8
__device__ __forceinline__ v16bf load_b16(const bf16* Wt, int ldn, int kt, int n0, int lane) {
  v16bf b;
  const int h = lane >> 4, n = n0 + (lane & 15);
#pragma unroll
  for (int s = 0; s < 16; ++s) {
    const int k = kt * 32 + (s < 8 ? 8 * h + s : 16 + 8 * h + (s - 8));
    b[s] = Wt[(size_t)k * ldn + n];
  }
  return b;
}

__device__ __forceinline__ v8f wmma_bf16(v16bf a, v16bf b, v8f c) {
  return __builtin_amdgcn_wmma_f32_16x16x32_bf16(false, a, false, b, (short)0, c,
                                                 false, false);
}

__device__ __forceinline__ float sigm(float x) { return 1.0f / (1.0f + __expf(-x)); }

// Accumulate r,z and one of (nx|nh) gate column groups over K.
// WGP-scope prefetch (locality 3) two K-iterations ahead so the B-fragments
// land in the near cache before their strided 16-bit element loads issue.
__device__ __forceinline__ void acc_gates(v8f& ar, v8f& az, v8f& an,
                                          const bf16* rowp, const bf16* Wt,
                                          int Ksz, int n0, int lane, int ks) {
  const int nkt = Ksz / 32;
  for (int kt = 0; kt < nkt; ++kt) {
    if (kt + 2 < nkt)
      __builtin_prefetch(Wt + (size_t)((kt + 2) * 32 + lane) * kH3 + n0, 0, 3);
    v16bf a = load_a16(rowp, kt * 32 + ks);
    ar = wmma_bf16(a, load_b16(Wt, kH3, kt, n0,          lane), ar);
    az = wmma_bf16(a, load_b16(Wt, kH3, kt, kH + n0,     lane), az);
    an = wmma_bf16(a, load_b16(Wt, kH3, kt, 2 * kH + n0, lane), an);
  }
}

// GRU gate math directly on C/D fragments (lane owns column j, rows mB..mB+7).
__device__ __forceinline__ void gru_update(v8f& hf, const v8f& ar, const v8f& az,
                                           const v8f& anx, const v8f& anh,
                                           const float* bih, const float* bhh, int j) {
  const float br = bih[j],          bhr = bhh[j];
  const float bz = bih[kH + j],     bhz = bhh[kH + j];
  const float bn = bih[2 * kH + j], bhn = bhh[2 * kH + j];
#pragma unroll
  for (int e = 0; e < 8; ++e) {
    float r = sigm(ar[e] + br + bhr);
    float z = sigm(az[e] + bz + bhz);
    float n = tanhf(anx[e] + bn + r * (anh[e] + bhn));
    hf[e] = (1.0f - z) * n + z * hf[e];
  }
}

__device__ __forceinline__ void store_h_lds(bf16* dst, const v8f& hf, int lane, int j) {
  const int mB = (lane >> 4) * 8;
#pragma unroll
  for (int e = 0; e < 8; ++e)
    dst[(mB + e) * kH + j] = (bf16)hf[e];
}

// ================= prep kernels =================
__global__ void k_transpose_bf16(bf16* __restrict__ dst, const float* __restrict__ src,
                                 int R, int C) {  // src RxC -> dst CxR (bf16)
  int i = blockIdx.x * blockDim.x + threadIdx.x;
  if (i >= R * C) return;
  int r = i / C, c = i - r * C;
  dst[(size_t)c * R + r] = (bf16)src[i];
}

__global__ void k_tobf16(bf16* __restrict__ dst, const float* __restrict__ src, int n) {
  int i = blockIdx.x * blockDim.x + threadIdx.x;
  if (i < n) dst[i] = (bf16)src[i];
}

__global__ void k_time_major_bf16(bf16* __restrict__ dst, const float* __restrict__ src) {
  // src [B][S][IN] -> dst [S][B][IN]
  int i = blockIdx.x * blockDim.x + threadIdx.x;
  if (i >= kB * kS * kIN) return;
  int f = i % kIN;
  int rest = i / kIN;
  int s = rest % kS;
  int b = rest / kS;
  dst[((size_t)s * kB + b) * kIN + f] = (bf16)src[i];
}

// ================= fused 2-layer GRU encoder scan =================
// grid = B/16 persistent WGs, 8 waves; each WG scans all S steps for 16 rows.
__global__ void __launch_bounds__(256, 1)
k_encoder(const bf16* __restrict__ xb,      // [S][B][IN] bf16
          const bf16* __restrict__ Wih0t, const bf16* __restrict__ Whh0t,  // [K][3H]
          const bf16* __restrict__ Wih1t, const bf16* __restrict__ Whh1t,
          const float* __restrict__ bih0, const float* __restrict__ bhh0,
          const float* __restrict__ bih1, const float* __restrict__ bhh1,
          float* __restrict__ hEnc)         // [2][B][H] f32
{
  __shared__ bf16 h0A[2][16 * kH];
  __shared__ bf16 h1A[2][16 * kH];
  const int tid  = threadIdx.x;
  const int lane = tid & 31;
  const int wave = tid >> 5;
  const int nl   = lane & 15;
  const int ks   = (lane >> 4) << 3;
  const int b0   = blockIdx.x * 16;

  for (int i = tid; i < 16 * kH; i += 256) {
    h0A[0][i] = (bf16)0.0f;
    h1A[0][i] = (bf16)0.0f;
  }
  v8f zf = {};
  v8f h0f[8], h1f[8];
#pragma unroll
  for (int t = 0; t < 8; ++t) { h0f[t] = zf; h1f[t] = zf; }
  __syncthreads();

  int p = 0;
  for (int s = 0; s < kS; ++s) {
    // ---- layer 0 (x: K=64 from global, h: K=1024 from LDS) ----
    const bf16* xrow  = xb + ((size_t)s * kB + b0 + nl) * kIN;
    const bf16* h0row = &h0A[p][nl * kH];
    for (int t = 0; t < 8; ++t) {
      const int n0 = (wave * 8 + t) * 16;
      v8f ar = zf, az = zf, anx = zf, anh = zf;
      acc_gates(ar, az, anx, xrow,  Wih0t, kIN, n0, lane, ks);
      acc_gates(ar, az, anh, h0row, Whh0t, kH,  n0, lane, ks);
      gru_update(h0f[t], ar, az, anx, anh, bih0, bhh0, n0 + nl);
      store_h_lds(h0A[p ^ 1], h0f[t], lane, n0 + nl);
    }
    __syncthreads();
    // ---- layer 1 (x = new h0) ----
    const bf16* y0row = &h0A[p ^ 1][nl * kH];
    const bf16* h1row = &h1A[p][nl * kH];
    for (int t = 0; t < 8; ++t) {
      const int n0 = (wave * 8 + t) * 16;
      v8f ar = zf, az = zf, anx = zf, anh = zf;
      acc_gates(ar, az, anx, y0row, Wih1t, kH, n0, lane, ks);
      acc_gates(ar, az, anh, h1row, Whh1t, kH, n0, lane, ks);
      gru_update(h1f[t], ar, az, anx, anh, bih1, bhh1, n0 + nl);
      store_h_lds(h1A[p ^ 1], h1f[t], lane, n0 + nl);
    }
    __syncthreads();
    p ^= 1;
  }

  for (int t = 0; t < 8; ++t) {
    const int j  = (wave * 8 + t) * 16 + nl;
    const int mB = (lane >> 4) * 8;
#pragma unroll
    for (int e = 0; e < 8; ++e) {
      hEnc[(size_t)(b0 + mB + e) * kH + j]                          = h0f[t][e];
      hEnc[(size_t)kB * kH + (size_t)(b0 + mB + e) * kH + j]        = h1f[t][e];
    }
  }
}

// ================= fused autoregressive decoder =================
__global__ void __launch_bounds__(256, 1)
k_decoder(const bf16* __restrict__ embB,    // [V][H] bf16
          const bf16* __restrict__ Wih0t, const bf16* __restrict__ Whh0t,
          const bf16* __restrict__ Wih1t, const bf16* __restrict__ Whh1t,
          const float* __restrict__ bih0, const float* __restrict__ bhh0,
          const float* __restrict__ bih1, const float* __restrict__ bhh1,
          const bf16* __restrict__ fcWt,    // [H][V] bf16
          const float* __restrict__ fcb,    // [V]
          const float* __restrict__ hEnc,   // [2][B][H]
          float* __restrict__ out)          // logits [B][T][V] then h_fin [2][B][H]
{
  __shared__ bf16  h0A[2][16 * kH];
  __shared__ bf16  h1A[2][16 * kH];
  __shared__ float logitsS[16 * kV];
  __shared__ int   tok[16];
  const int tid  = threadIdx.x;
  const int lane = tid & 31;
  const int wave = tid >> 5;
  const int nl   = lane & 15;
  const int ks   = (lane >> 4) << 3;
  const int b0   = blockIdx.x * 16;

  for (int i = tid; i < 16 * kH; i += 256) {
    int m = i >> 10, j = i & (kH - 1);
    h0A[0][i] = (bf16)hEnc[(size_t)(b0 + m) * kH + j];
    h1A[0][i] = (bf16)hEnc[(size_t)kB * kH + (size_t)(b0 + m) * kH + j];
  }
  if (tid < 16) tok[tid] = 0;  // start token 0 (pad, zero embedding)
  v8f zf = {};
  v8f h0f[8], h1f[8];
  for (int t = 0; t < 8; ++t) {
    const int j  = (wave * 8 + t) * 16 + nl;
    const int mB = (lane >> 4) * 8;
#pragma unroll
    for (int e = 0; e < 8; ++e) {
      h0f[t][e] = hEnc[(size_t)(b0 + mB + e) * kH + j];
      h1f[t][e] = hEnc[(size_t)kB * kH + (size_t)(b0 + mB + e) * kH + j];
    }
  }
  __syncthreads();

  int p = 0;
  for (int step = 0; step < kT; ++step) {
    // ---- layer 0: input = embedding row of this lane's batch row's token ----
    const bf16* xrow  = embB + (size_t)tok[nl] * kH;
    const bf16* h0row = &h0A[p][nl * kH];
    for (int t = 0; t < 8; ++t) {
      const int n0 = (wave * 8 + t) * 16;
      v8f ar = zf, az = zf, anx = zf, anh = zf;
      acc_gates(ar, az, anx, xrow,  Wih0t, kH, n0, lane, ks);
      acc_gates(ar, az, anh, h0row, Whh0t, kH, n0, lane, ks);
      gru_update(h0f[t], ar, az, anx, anh, bih0, bhh0, n0 + nl);
      store_h_lds(h0A[p ^ 1], h0f[t], lane, n0 + nl);
    }
    __syncthreads();
    // ---- layer 1 ----
    const bf16* y0row = &h0A[p ^ 1][nl * kH];
    const bf16* h1row = &h1A[p][nl * kH];
    for (int t = 0; t < 8; ++t) {
      const int n0 = (wave * 8 + t) * 16;
      v8f ar = zf, az = zf, anx = zf, anh = zf;
      acc_gates(ar, az, anx, y0row, Wih1t, kH, n0, lane, ks);
      acc_gates(ar, az, anh, h1row, Whh1t, kH, n0, lane, ks);
      gru_update(h1f[t], ar, az, anx, anh, bih1, bhh1, n0 + nl);
      store_h_lds(h1A[p ^ 1], h1f[t], lane, n0 + nl);
    }
    __syncthreads();
    // ---- FC head: logits = h1_new @ fcW^T + b (WMMA, K=1024) ----
    const bf16* h1new = &h1A[p ^ 1][nl * kH];
    for (int q = 0; q < 4; ++q) {
      const int n0 = (wave * 4 + q) * 16;
      v8f acc = zf;
      for (int kt = 0; kt < kH / 32; ++kt) {
        if (kt + 2 < kH / 32)
          __builtin_prefetch(fcWt + (size_t)((kt + 2) * 32 + lane) * kV + n0, 0, 3);
        acc = wmma_bf16(load_a16(h1new, kt * 32 + ks),
                        load_b16(fcWt, kV, kt, n0, lane), acc);
      }
      const float bias = fcb[n0 + nl];
      const int mB = (lane >> 4) * 8;
#pragma unroll
      for (int e = 0; e < 8; ++e)
        logitsS[(mB + e) * kV + n0 + nl] = acc[e] + bias;
    }
    __syncthreads();
    // ---- emit logits + greedy argmax feedback ----
    for (int i = tid; i < 16 * kV; i += 256) {
      int m = i >> 9, v = i & (kV - 1);
      out[((size_t)(b0 + m) * kT + step) * kV + v] = logitsS[i];
    }
    if (tid < 16) {
      const float* row = &logitsS[tid * kV];
      float best = row[0];
      int bi = 0;
      for (int v = 1; v < kV; ++v) {
        float x = row[v];
        if (x > best) { best = x; bi = v; }
      }
      tok[tid] = bi;
    }
    __syncthreads();
    p ^= 1;
  }

  const size_t hoff = (size_t)kB * kT * kV;
  for (int t = 0; t < 8; ++t) {
    const int j  = (wave * 8 + t) * 16 + nl;
    const int mB = (lane >> 4) * 8;
#pragma unroll
    for (int e = 0; e < 8; ++e) {
      out[hoff + (size_t)(b0 + mB + e) * kH + j]                   = h0f[t][e];
      out[hoff + (size_t)kB * kH + (size_t)(b0 + mB + e) * kH + j] = h1f[t][e];
    }
  }
}

// ================= host =================
extern "C" void kernel_launch(void* const* d_in, const int* in_sizes, int n_in,
                              void* d_out, int out_size, void* d_ws, size_t ws_size,
                              hipStream_t stream) {
  (void)in_sizes; (void)n_in; (void)out_size; (void)ws_size;
  const float* x     = (const float*)d_in[0];
  const float* emb   = (const float*)d_in[1];
  const float* eWih0 = (const float*)d_in[2];
  const float* eWhh0 = (const float*)d_in[3];
  const float* ebih0 = (const float*)d_in[4];
  const float* ebhh0 = (const float*)d_in[5];
  const float* eWih1 = (const float*)d_in[6];
  const float* eWhh1 = (const float*)d_in[7];
  const float* ebih1 = (const float*)d_in[8];
  const float* ebhh1 = (const float*)d_in[9];
  const float* dWih  = (const float*)d_in[10];  // (2,3H,H)
  const float* dWhh  = (const float*)d_in[11];
  const float* dbih  = (const float*)d_in[12];  // (2,3H)
  const float* dbhh  = (const float*)d_in[13];
  const float* fcW   = (const float*)d_in[14];  // (V,H)
  const float* fcb   = (const float*)d_in[15];
  float* out = (float*)d_out;

  char* w = (char*)d_ws;
  auto alloc = [&](size_t bytes) -> void* {
    void* p = (void*)w;
    w += (bytes + 255) & ~(size_t)255;
    return p;
  };
  bf16* xb     = (bf16*)alloc((size_t)kS * kB * kIN * sizeof(bf16));
  bf16* Wih0t  = (bf16*)alloc((size_t)kIN * kH3 * sizeof(bf16));
  bf16* Whh0t  = (bf16*)alloc((size_t)kH * kH3 * sizeof(bf16));
  bf16* Wih1t  = (bf16*)alloc((size_t)kH * kH3 * sizeof(bf16));
  bf16* Whh1t  = (bf16*)alloc((size_t)kH * kH3 * sizeof(bf16));
  bf16* dWih0t = (bf16*)alloc((size_t)kH * kH3 * sizeof(bf16));
  bf16* dWhh0t = (bf16*)alloc((size_t)kH * kH3 * sizeof(bf16));
  bf16* dWih1t = (bf16*)alloc((size_t)kH * kH3 * sizeof(bf16));
  bf16* dWhh1t = (bf16*)alloc((size_t)kH * kH3 * sizeof(bf16));
  bf16* fcWt   = (bf16*)alloc((size_t)kH * kV * sizeof(bf16));
  bf16* embB   = (bf16*)alloc((size_t)kV * kH * sizeof(bf16));
  float* hEnc  = (float*)alloc((size_t)2 * kB * kH * sizeof(float));

  const int TPB = 256;
  auto NB = [](long n) { return (int)((n + 255) / 256); };

  k_time_major_bf16<<<NB((long)kB * kS * kIN), TPB, 0, stream>>>(xb, x);
  k_transpose_bf16<<<NB((long)kH3 * kIN), TPB, 0, stream>>>(Wih0t, eWih0, kH3, kIN);
  k_transpose_bf16<<<NB((long)kH3 * kH), TPB, 0, stream>>>(Whh0t, eWhh0, kH3, kH);
  k_transpose_bf16<<<NB((long)kH3 * kH), TPB, 0, stream>>>(Wih1t, eWih1, kH3, kH);
  k_transpose_bf16<<<NB((long)kH3 * kH), TPB, 0, stream>>>(Whh1t, eWhh1, kH3, kH);
  k_transpose_bf16<<<NB((long)kH3 * kH), TPB, 0, stream>>>(dWih0t, dWih, kH3, kH);
  k_transpose_bf16<<<NB((long)kH3 * kH), TPB, 0, stream>>>(dWhh0t, dWhh, kH3, kH);
  k_transpose_bf16<<<NB((long)kH3 * kH), TPB, 0, stream>>>(dWih1t, dWih + (size_t)kH3 * kH, kH3, kH);
  k_transpose_bf16<<<NB((long)kH3 * kH), TPB, 0, stream>>>(dWhh1t, dWhh + (size_t)kH3 * kH, kH3, kH);
  k_transpose_bf16<<<NB((long)kV * kH), TPB, 0, stream>>>(fcWt, fcW, kV, kH);
  k_tobf16<<<NB((long)kV * kH), TPB, 0, stream>>>(embB, emb, kV * kH);

  k_encoder<<<kB / 16, 256, 0, stream>>>(xb, Wih0t, Whh0t, Wih1t, Whh1t,
                                         ebih0, ebhh0, ebih1, ebhh1, hEnc);
  k_decoder<<<kB / 16, 256, 0, stream>>>(embB, dWih0t, dWhh0t, dWih1t, dWhh1t,
                                         dbih, dbhh, dbih + kH3, dbhh + kH3,
                                         fcWt, fcb, hEnc, out);
}
